// DeeperGCN_43181601194036
// MI455X (gfx1250) — matrix operand order repaired and checked by reading, not compiled, not run on hardware
//
#include <hip/hip_runtime.h>
#include <hip/hip_bf16.h>

#define HDIM 256
#define ATOM_NF 9
#define ATOM_VOCAB 64
#define BOND_NF 3
#define BOND_VOCAB 8
#define BN_EPS 1e-5f
#define MSG_EPS 1e-7f

typedef __attribute__((ext_vector_type(16))) __bf16 v16bf;
typedef __attribute__((ext_vector_type(8)))  float  v8f;

// ---------------------------------------------------------------------------
// Preconvert W[256,256] f32 -> bf16 (done once per GEMM; W is L2-resident and
// shared by ~4.7k blocks, so convert-once instead of convert-per-block).
// ---------------------------------------------------------------------------
__global__ __launch_bounds__(256) void k_w_to_bf16(const float* __restrict__ w,
                                                   __bf16* __restrict__ out) {
  const int i = (blockIdx.x * 256 + threadIdx.x) * 4;   // HDIM*HDIM/1024 blocks
  out[i + 0] = (__bf16)w[i + 0];
  out[i + 1] = (__bf16)w[i + 1];
  out[i + 2] = (__bf16)w[i + 2];
  out[i + 3] = (__bf16)w[i + 3];
}

// ---------------------------------------------------------------------------
// WMMA GEMM:  C[M,256] = (A0 (+A1)) @ Wbf16[256,256] + bias (+resid)
// A converted f32->bf16 inline (fused h2+agg add), f32 accumulate.
// Wave = 16x64 output tile (4 WMMA accs), block = 8 waves = 128x64.
// A rows are CLAMPED (not branched) so the K-loop has no EXEC manipulation;
// out-of-range rows are masked only at the store epilogue.
// ---------------------------------------------------------------------------
template <bool HAS_A1, bool HAS_RESID>
__global__ __launch_bounds__(256) void k_gemm_wmma(
    const float* __restrict__ A0, const float* __restrict__ A1,
    const __bf16* __restrict__ Wb, const float* __restrict__ bias,
    const float* __restrict__ resid, float* __restrict__ C, int M)
{
  const int lane    = threadIdx.x & 31;
  const int wave    = threadIdx.x >> 5;
  const int m0      = blockIdx.x * 128 + wave * 16;
  const int n0      = blockIdx.y * 64;
  if (m0 >= M) return;                 // wave-uniform: EXEC stays all-ones

  const int halfsel = lane >> 4;       // 0 -> lanes 0-15, 1 -> lanes 16-31
  const int mlane   = lane & 15;
  int arow = m0 + mlane;
  if (arow > M - 1) arow = M - 1;      // clamp: always-legal load, masked at store

  const float* ap = A0 + (size_t)arow * HDIM;
  const float* aq = HAS_A1 ? (A1 + (size_t)arow * HDIM) : nullptr;

  v8f acc[4];
  #pragma unroll
  for (int t = 0; t < 4; ++t) acc[t] = (v8f){};

  for (int kk = 0; kk < HDIM; kk += 32) {
    // ---- A fragment (16x32 bf16): two contiguous 8-element runs per lane
    const int kb0 = kk + (halfsel ? 8 : 0);
    const int kb1 = kb0 + 16;
    __builtin_prefetch(ap + kb0 + 32, 0, 1);          // next K-chunk
    v16bf afrag;
    if (HAS_A1) {
      #pragma unroll
      for (int j = 0; j < 8; ++j) {
        afrag[j]     = (__bf16)(ap[kb0 + j] + aq[kb0 + j]);
        afrag[8 + j] = (__bf16)(ap[kb1 + j] + aq[kb1 + j]);
      }
    } else {
      #pragma unroll
      for (int j = 0; j < 8; ++j) {
        afrag[j]     = (__bf16)ap[kb0 + j];
        afrag[8 + j] = (__bf16)ap[kb1 + j];
      }
    }
    // ---- B fragments (32x16 bf16): lane holds col n=lane%16, K rows kb0..+7 and kb1..+7,
    // already bf16 in memory -> strided gather, no cvt.
    const __bf16* wp = Wb + (size_t)kb0 * HDIM + n0 + mlane;
    #pragma unroll
    for (int t = 0; t < 4; ++t) {
      v16bf bfrag;
      #pragma unroll
      for (int j = 0; j < 8; ++j) {
        bfrag[j]     = wp[(size_t)j * HDIM + t * 16];
        bfrag[8 + j] = wp[(size_t)(16 + j) * HDIM + t * 16];
      }
      acc[t] = __builtin_amdgcn_wmma_f32_16x16x32_bf16(
          false, afrag, false, bfrag, (short)0, acc[t], false, false);
    }
  }

  // ---- epilogue: C layout VGPR r -> row r (lanes 0-15) / r+8 (lanes 16-31)
  #pragma unroll
  for (int t = 0; t < 4; ++t) {
    const int col = n0 + t * 16 + mlane;
    const float bc = bias[col];
    #pragma unroll
    for (int r = 0; r < 8; ++r) {
      const int row = m0 + r + halfsel * 8;
      if (row < M) {
        float v = acc[t][r] + bc;
        if (HAS_RESID) v += resid[(size_t)row * HDIM + col];
        C[(size_t)row * HDIM + col] = v;
      }
    }
  }
}

// ---------------------------------------------------------------------------
// Supporting elementwise / scatter / reduction kernels
// ---------------------------------------------------------------------------
__global__ __launch_bounds__(256) void k_zero(float* __restrict__ p, long long n) {
  long long i = (long long)blockIdx.x * 256 + threadIdx.x;
  if (i < n) p[i] = 0.0f;
}

// out[r,:] = (add_row? add_row : 0) + sum_f emb[f, feats[r,f], :]
__global__ __launch_bounds__(256) void k_encode(const int* __restrict__ feats,
                                                const float* __restrict__ emb,
                                                const float* __restrict__ add_row,
                                                float* __restrict__ out,
                                                int nf, int vocab) {
  const int r = blockIdx.x, c = threadIdx.x;
  float s = add_row ? add_row[c] : 0.0f;
  for (int f = 0; f < nf; ++f) {
    const int v = feats[r * nf + f];
    s += emb[((size_t)f * vocab + v) * HDIM + c];
  }
  out[(size_t)r * HDIM + c] = s;
}

__global__ __launch_bounds__(256) void k_broadcast_row(const float* __restrict__ row,
                                                       float* __restrict__ out) {
  out[(size_t)blockIdx.x * HDIM + threadIdx.x] = row[threadIdx.x];
}

// msg = relu(h[src] + ee) + eps ; agg[dst] += msg   (one block per edge)
__global__ __launch_bounds__(256) void k_edge_msg(const float* __restrict__ h,
                                                  const float* __restrict__ ee,
                                                  const int* __restrict__ src,
                                                  const int* __restrict__ dst,
                                                  float* __restrict__ agg) {
  const int e = blockIdx.x, c = threadIdx.x;
  const int s = src[e], d = dst[e];
  const float m = fmaxf(h[(size_t)s * HDIM + c] + ee[(size_t)e * HDIM + c], 0.0f) + MSG_EPS;
  atomicAdd(&agg[(size_t)d * HDIM + c], m);
}

__global__ __launch_bounds__(256) void k_seg_pool(const float* __restrict__ x,
                                                  const int* __restrict__ seg,
                                                  float* __restrict__ out) {
  const int r = blockIdx.x, c = threadIdx.x;
  atomicAdd(&out[(size_t)seg[r] * HDIM + c], x[(size_t)r * HDIM + c]);
}

__global__ __launch_bounds__(256) void k_ew_add(float* __restrict__ y,
                                                const float* __restrict__ a, long long n) {
  long long i = (long long)blockIdx.x * 256 + threadIdx.x;
  if (i < n) y[i] += a[i];
}

__global__ __launch_bounds__(256) void k_add_gather(float* __restrict__ y,
                                                    const float* __restrict__ vn,
                                                    const int* __restrict__ seg) {
  const int r = blockIdx.x, c = threadIdx.x;
  y[(size_t)r * HDIM + c] += vn[(size_t)seg[r] * HDIM + c];
}

// Per-channel sum / sum-of-squares partials, atomically combined.
__global__ __launch_bounds__(256) void k_bn_stats(const float* __restrict__ x, int rows,
                                                  int rpb, float* __restrict__ sums,
                                                  float* __restrict__ sq) {
  const int c = threadIdx.x;
  int r0 = blockIdx.x * rpb;
  int r1 = r0 + rpb; if (r1 > rows) r1 = rows;
  float s = 0.f, q = 0.f;
  for (int r = r0; r < r1; ++r) {
    const float v = x[(size_t)r * HDIM + c];
    s += v; q += v * v;
  }
  atomicAdd(&sums[c], s);
  atomicAdd(&sq[c], q);
}

// Fold mean/var/gamma/beta into y = x*scale + shift
__global__ __launch_bounds__(256) void k_bn_finalize(const float* __restrict__ sums,
                                                     const float* __restrict__ sq, int rows,
                                                     const float* __restrict__ gamma,
                                                     const float* __restrict__ beta,
                                                     float* __restrict__ scale,
                                                     float* __restrict__ shift) {
  const int c = threadIdx.x;
  const float inv = 1.0f / (float)rows;
  const float mu  = sums[c] * inv;
  const float var = sq[c] * inv - mu * mu;
  const float sc  = rsqrtf(var + BN_EPS) * gamma[c];
  scale[c] = sc;
  shift[c] = beta[c] - mu * sc;
}

__global__ __launch_bounds__(256) void k_bn_apply(const float* __restrict__ x,
                                                  const float* __restrict__ scale,
                                                  const float* __restrict__ shift,
                                                  float* __restrict__ y,
                                                  long long n, int do_relu) {
  long long i = (long long)blockIdx.x * 256 + threadIdx.x;
  if (i >= n) return;
  const int c = (int)(i & (HDIM - 1));
  float v = x[i] * scale[c] + shift[c];
  if (do_relu) v = fmaxf(v, 0.0f);
  y[i] = v;
}

// ---------------------------------------------------------------------------
// Orchestration
// ---------------------------------------------------------------------------
extern "C" void kernel_launch(void* const* d_in, const int* in_sizes, int n_in,
                              void* d_out, int out_size, void* d_ws, size_t ws_size,
                              hipStream_t stream) {
  (void)n_in; (void)ws_size;
  const int*   x_feats   = (const int*)  d_in[0];
  const int*   edge_idx  = (const int*)  d_in[1];
  const int*   edge_attr = (const int*)  d_in[2];
  const int*   batch     = (const int*)  d_in[3];
  const float* atom_emb  = (const float*)d_in[4];
  const float* bond_emb  = (const float*)d_in[5];
  const float* vn_weight = (const float*)d_in[6];
  const float* conv_w    = (const float*)d_in[7];
  const float* conv_b    = (const float*)d_in[8];
  const float* norm_g    = (const float*)d_in[9];
  const float* norm_be   = (const float*)d_in[10];
  const float* vn_w      = (const float*)d_in[11];
  const float* vn_bias   = (const float*)d_in[12];
  const float* vn_g      = (const float*)d_in[13];
  const float* vn_be     = (const float*)d_in[14];

  const int N = in_sizes[3];
  const int E = in_sizes[1] / 2;
  const int G = out_size / HDIM;
  const int L = in_sizes[7] / (HDIM * HDIM);   // 7

  const int* src = edge_idx;
  const int* dst = edge_idx + E;

  // workspace layout (f32 units)
  float* ws = (float*)d_ws;
  const size_t NH = (size_t)N * HDIM;
  const size_t EH = (size_t)E * HDIM;
  const size_t GH = (size_t)G * HDIM;
  float* hA    = ws; ws += NH;
  float* hB    = ws; ws += NH;
  float* h2    = ws; ws += NH;
  float* agg   = ws; ws += NH;
  float* ee    = ws; ws += EH;
  float* vn    = ws; ws += GH;
  float* vnt   = ws; ws += GH;
  float* vtmp  = ws; ws += GH;
  float* sums  = ws; ws += HDIM;
  float* sq    = ws; ws += HDIM;
  float* scale = ws; ws += HDIM;
  float* shift = ws; ws += HDIM;
  __bf16* wbf  = (__bf16*)ws; ws += (HDIM * HDIM) / 2;   // 64K bf16 staging for W

  auto zero = [&](float* p, long long n) {
    k_zero<<<dim3((unsigned)((n + 255) / 256)), 256, 0, stream>>>(p, n);
  };
  auto bn = [&](const float* xin, int rows, const float* g, const float* b,
                float* yout, int relu) {
    zero(sums, HDIM); zero(sq, HDIM);
    const int rpb = 256;
    k_bn_stats<<<dim3((unsigned)((rows + rpb - 1) / rpb)), 256, 0, stream>>>(
        xin, rows, rpb, sums, sq);
    k_bn_finalize<<<1, 256, 0, stream>>>(sums, sq, rows, g, b, scale, shift);
    const long long n = (long long)rows * HDIM;
    k_bn_apply<<<dim3((unsigned)((n + 255) / 256)), 256, 0, stream>>>(
        xin, scale, shift, yout, n, relu);
  };
  auto gemm = [&](const float* A0, const float* A1, const float* W, const float* bias,
                  const float* resid, float* Cout, int M) {
    k_w_to_bf16<<<dim3((HDIM * HDIM) / 1024), 256, 0, stream>>>(W, wbf);
    dim3 grid((unsigned)((M + 127) / 128), HDIM / 64);
    if (A1) {
      if (resid)
        k_gemm_wmma<true, true><<<grid, 256, 0, stream>>>(A0, A1, wbf, bias, resid, Cout, M);
      else
        k_gemm_wmma<true, false><<<grid, 256, 0, stream>>>(A0, A1, wbf, bias, nullptr, Cout, M);
    } else {
      k_gemm_wmma<false, false><<<grid, 256, 0, stream>>>(A0, nullptr, wbf, bias, nullptr, Cout, M);
    }
  };

  // encoders + virtual-node init (h gets vn_weight row added: vn[batch] broadcast)
  k_encode<<<N, 256, 0, stream>>>(x_feats, atom_emb, vn_weight, hA, ATOM_NF, ATOM_VOCAB);
  k_encode<<<E, 256, 0, stream>>>(edge_attr, bond_emb, nullptr, ee, BOND_NF, BOND_VOCAB);
  k_broadcast_row<<<G, 256, 0, stream>>>(vn_weight, vn);

  // layer 0: GENConv
  zero(agg, (long long)NH);
  k_edge_msg<<<E, 256, 0, stream>>>(hA, ee, src, dst, agg);
  gemm(hA, agg, conv_w, conv_b, nullptr, hB, N);
  float* h = hB; float* hOther = hA;

  for (int l = 1; l < L; ++l) {
    // BN -> ReLU
    bn(h, N, norm_g + (size_t)(l - 1) * HDIM, norm_be + (size_t)(l - 1) * HDIM, h2, 1);

    // virtual node: pool + residual, 2x (Linear -> BN -> ReLU)
    zero(vnt, (long long)GH);
    k_seg_pool<<<N, 256, 0, stream>>>(h2, batch, vnt);
    k_ew_add<<<dim3((unsigned)((GH + 255) / 256)), 256, 0, stream>>>(vnt, vn, (long long)GH);
    for (int j = 0; j < 2; ++j) {
      const size_t o = ((size_t)(l - 1) * 2 + j);
      gemm(vnt, nullptr, vn_w + o * HDIM * HDIM, vn_bias + o * HDIM, nullptr, vtmp, G);
      bn(vtmp, G, vn_g + o * HDIM, vn_be + o * HDIM, vnt, 1);
    }
    { float* t = vn; vn = vnt; vnt = t; }

    // h2 += vn[batch]; GENConv; residual fused into GEMM epilogue
    k_add_gather<<<N, 256, 0, stream>>>(h2, vn, batch);
    zero(agg, (long long)NH);
    k_edge_msg<<<E, 256, 0, stream>>>(h2, ee, src, dst, agg);
    gemm(h2, agg, conv_w + (size_t)l * HDIM * HDIM, conv_b + (size_t)l * HDIM,
         h, hOther, N);
    { float* t = h; h = hOther; hOther = t; }
  }

  // final BN (no relu) + global_add_pool
  bn(h, N, norm_g + (size_t)(L - 1) * HDIM, norm_be + (size_t)(L - 1) * HDIM, h2, 0);
  zero((float*)d_out, (long long)G * HDIM);
  k_seg_pool<<<N, 256, 0, stream>>>(h2, batch, (float*)d_out);
}
